// MultiheadEinsum_6485400617574
// MI455X (gfx1250) — matrix operand order repaired and checked by reading, not compiled, not run on hardware
//
#include <hip/hip_runtime.h>

#define B_ 4
#define S_ 2048
#define E_ 1024
#define H_ 16
#define D_ 64
#define TD_ 192  // 3*D

typedef __attribute__((ext_vector_type(16))) _Float16 v16h;
typedef __attribute__((ext_vector_type(8)))  _Float16 v8h;
typedef __attribute__((ext_vector_type(8)))  float    v8f;

union HV16 { v16h v; v8h h[2]; };

static __device__ __forceinline__ v16h mk16(v8h lo, v8h hi) {
  HV16 u; u.h[0] = lo; u.h[1] = hi; return u.v;
}
static __device__ __forceinline__ v8h ld8h(const _Float16* p) {
  return *(const v8h*)p;
}
static __device__ __forceinline__ v16h ld16h(const _Float16* p) {
  return *(const v16h*)p;
}
static __device__ __forceinline__ v8h cvt8(const float* p) {
  v8h r;
#pragma unroll
  for (int i = 0; i < 8; ++i) r[i] = (_Float16)p[i];
  return r;
}
static __device__ __forceinline__ v8f wmma_f16(v16h a, v16h b, v8f c) {
  return __builtin_amdgcn_wmma_f32_16x16x32_f16(false, a, false, b, (short)0, c,
                                                false, false);
}

// ---------------------------------------------------------------- fp32 -> f16
__global__ void cvt_f32_to_f16(const float* __restrict__ src,
                               _Float16* __restrict__ dst, int n) {
  int i = blockIdx.x * blockDim.x + threadIdx.x;
  if (i < n) dst[i] = (_Float16)src[i];
}

// ------------------------------------------------------------- QKV projection
// One wave per (b,h, 16-row s-tile). Computes 16x192 = X(16x64) @ W^T(64x192).
// Q,K stored [BH,S,D]; V stored transposed [BH,D,S] for the P@V operand.
__global__ __launch_bounds__(32) void qkv_kernel(
    const float* __restrict__ x, const _Float16* __restrict__ Wq,
    _Float16* __restrict__ Q, _Float16* __restrict__ K,
    _Float16* __restrict__ Vt) {
  const int bh = blockIdx.x;
  const int b = bh >> 4, h = bh & 15;
  const int s0 = blockIdx.y * 16;
  const int l = threadIdx.x;
  const int n = l & 15;
  const int hf = l >> 4;
  const int kc = hf * 8;  // unused; A-operand K chunks start at 16*hf

  // A operand: rows of x (f32 -> f16), K = d in [0,64)
  const float* xrow = x + ((size_t)(b * S_ + s0 + n) * H_ + h) * D_;
  const v16h a0 = mk16(cvt8(xrow + 16 * hf), cvt8(xrow + 16 * hf + 8));
  const v16h a1 = mk16(cvt8(xrow + 32 + 16 * hf), cvt8(xrow + 40 + 16 * hf));

#pragma unroll
  for (int nt = 0; nt < 12; ++nt) {
    // B operand: lane holds output column e, 16 contiguous K per lane
    const _Float16* wrow = Wq + ((size_t)h * TD_ + nt * 16 + n) * D_;
    v8f c = {};
    c = wmma_f16(a0, ld16h(wrow + 16 * hf), c);
    c = wmma_f16(a1, ld16h(wrow + 32 + 16 * hf), c);
    if (nt < 8) {  // Q or K: [BH, S, D]
      _Float16* dst = (nt < 4) ? Q : K;
      const int d = (nt & 3) * 16 + n;
#pragma unroll
      for (int r = 0; r < 8; ++r) {
        const int m = r + 8 * hf;
        dst[((size_t)bh * S_ + s0 + m) * D_ + d] = (_Float16)c[r];
      }
    } else {  // V transposed: [BH, D, S]; consecutive r -> consecutive s
      const int d = (nt - 8) * 16 + n;
      v8h o;
#pragma unroll
      for (int r = 0; r < 8; ++r) o[r] = (_Float16)c[r];
      *(v8h*)(Vt + ((size_t)bh * D_ + d) * S_ + s0 + 8 * hf) = o;
    }
  }
}

// ------------------------------------------------------------ flash attention
// 4 waves/block; each wave owns a 16-query tile, streams 32-key blocks.
__global__ __launch_bounds__(128) void attn_kernel(
    const _Float16* __restrict__ Q, const _Float16* __restrict__ K,
    const _Float16* __restrict__ Vt, _Float16* __restrict__ O) {
  __shared__ __align__(16) _Float16 pbuf[4][16][32];  // per-wave P bounce
  const int bh = blockIdx.x;
  const int w = threadIdx.x >> 5;
  const int l = threadIdx.x & 31;
  const int n = l & 15;
  const int hf = l >> 4;
  const int kc = hf * 8;
  const int s0 = blockIdx.y * 64 + w * 16;

  // Q tile as A operand, with 1/sqrt(D)=0.125 folded in (exact, power of 2)
  const _Float16* qrow = Q + ((size_t)bh * S_ + s0 + n) * D_;
  v16h aq0 = mk16(ld8h(qrow + 16 * hf), ld8h(qrow + 16 * hf + 8));
  v16h aq1 = mk16(ld8h(qrow + 32 + 16 * hf), ld8h(qrow + 40 + 16 * hf));
  const _Float16 sc = (_Float16)0.125f;
#pragma unroll
  for (int i = 0; i < 16; ++i) { aq0[i] *= sc; aq1[i] *= sc; }

  float mrun[8], lrun[8];
  v8f acc[4];
#pragma unroll
  for (int r = 0; r < 8; ++r) { mrun[r] = -3.0e38f; lrun[r] = 0.0f; }
#pragma unroll
  for (int dt = 0; dt < 4; ++dt) { v8f z = {}; acc[dt] = z; }

  const _Float16* Kb = K + (size_t)bh * S_ * D_;
  const _Float16* Vb = Vt + (size_t)bh * D_ * S_;

  for (int j0 = 0; j0 < S_; j0 += 32) {
    if (j0 + 32 < S_)
      __builtin_prefetch(Kb + (size_t)(j0 + 32 + n) * D_, 0, 0);

    // scores: S = Qs @ K^T ; B operand = K rows (contiguous 16 K per lane)
    const _Float16* kr0 = Kb + (size_t)(j0 + n) * D_;
    const _Float16* kr1 = Kb + (size_t)(j0 + 16 + n) * D_;
    v8f sa = {}, sb = {};
    sa = wmma_f16(aq0, ld16h(kr0 + 16 * hf), sa);
    sa = wmma_f16(aq1, ld16h(kr0 + 32 + 16 * hf), sa);
    sb = wmma_f16(aq0, ld16h(kr1 + 16 * hf), sb);
    sb = wmma_f16(aq1, ld16h(kr1 + 32 + 16 * hf), sb);

    // online softmax (row m = r + 8*hf lives across the lane's 16-lane half)
    float pa[8], pb[8], alpha[8];
#pragma unroll
    for (int r = 0; r < 8; ++r) {
      float mx = fmaxf(sa[r], sb[r]);
#pragma unroll
      for (int msk = 1; msk < 16; msk <<= 1)
        mx = fmaxf(mx, __shfl_xor(mx, msk, 32));
      const float mnew = fmaxf(mrun[r], mx);
      alpha[r] = __expf(mrun[r] - mnew);
      pa[r] = __expf(sa[r] - mnew);
      pb[r] = __expf(sb[r] - mnew);
      float rs = pa[r] + pb[r];
#pragma unroll
      for (int msk = 1; msk < 16; msk <<= 1) rs += __shfl_xor(rs, msk, 32);
      lrun[r] = lrun[r] * alpha[r] + rs;
      mrun[r] = mnew;
    }
    v8f av;
#pragma unroll
    for (int r = 0; r < 8; ++r) av[r] = alpha[r];
#pragma unroll
    for (int dt = 0; dt < 4; ++dt) acc[dt] *= av;

    // P: C-layout -> LDS -> A-layout (in-order per-wave LDS, no barrier)
#pragma unroll
    for (int r = 0; r < 8; ++r) {
      const int m = r + 8 * hf;
      pbuf[w][m][n] = (_Float16)pa[r];
      pbuf[w][m][16 + n] = (_Float16)pb[r];
    }
    asm volatile("s_wait_dscnt 0" ::: "memory");
    const v8h* prow = (const v8h*)&pbuf[w][n][0];
    const v16h pA = mk16(prow[2 * hf], prow[2 * hf + 1]);
    asm volatile("" ::: "memory");  // keep reads ahead of next iter's stores

    // O += P @ V ; B operand = rows of V^T (contiguous 16 j per lane)
#pragma unroll
    for (int dt = 0; dt < 4; ++dt) {
      const _Float16* vrow = Vb + (size_t)(dt * 16 + n) * S_ + j0;
      acc[dt] = wmma_f16(pA, ld16h(vrow + 16 * hf), acc[dt]);
    }
  }

  v8f inv;
#pragma unroll
  for (int r = 0; r < 8; ++r) inv[r] = 1.0f / lrun[r];
#pragma unroll
  for (int dt = 0; dt < 4; ++dt) {
    const v8f o = acc[dt] * inv;
#pragma unroll
    for (int r = 0; r < 8; ++r) {
      const int m = r + 8 * hf;
      O[((size_t)bh * S_ + s0 + m) * D_ + dt * 16 + n] = (_Float16)o[r];
    }
  }
}

// ----------------------------------------------------------- output proj GEMM
// out[row, e] = sum_j O_flat[row, j] * Wout[e, j]; row = b*S+s, j = h*64+d.
__global__ __launch_bounds__(32) void proj_kernel(
    const _Float16* __restrict__ O, const _Float16* __restrict__ Wo,
    float* __restrict__ out) {
  const int mt = blockIdx.x;       // 16-row tile of B*S
  const int n0 = blockIdx.y * 64;  // 64-col group of E
  const int l = threadIdx.x;
  const int n = l & 15;
  const int hf = l >> 4;
  const int kc = hf * 8;
  const int row = mt * 16 + n;
  const int bb = row >> 11;        // / S_
  const int s = row & (S_ - 1);

  v8f c[4];
#pragma unroll
  for (int nt = 0; nt < 4; ++nt) { v8f z = {}; c[nt] = z; }

  for (int kb = 0; kb < E_; kb += 32) {
    const int j1 = kb + 16 * hf;   // each 8-chunk stays within one head block
    const int j2 = kb + 16 * hf + 8;
    const _Float16* ap1 =
        O + ((size_t)(bb * H_ + (j1 >> 6)) * S_ + s) * D_ + (j1 & 63);
    const _Float16* ap2 =
        O + ((size_t)(bb * H_ + (j2 >> 6)) * S_ + s) * D_ + (j2 & 63);
    const v16h a = mk16(ld8h(ap1), ld8h(ap2));
#pragma unroll
    for (int nt = 0; nt < 4; ++nt) {
      const _Float16* wrow = Wo + (size_t)(n0 + nt * 16 + n) * E_ + kb;
      c[nt] = wmma_f16(a, ld16h(wrow + 16 * hf), c[nt]);
    }
  }
#pragma unroll
  for (int nt = 0; nt < 4; ++nt) {
#pragma unroll
    for (int r = 0; r < 8; ++r) {
      const int m = r + 8 * hf;
      out[(size_t)(mt * 16 + m) * E_ + n0 + nt * 16 + n] = c[nt][r];
    }
  }
}

// -------------------------------------------------------------------- launch
extern "C" void kernel_launch(void* const* d_in, const int* in_sizes, int n_in,
                              void* d_out, int out_size, void* d_ws,
                              size_t ws_size, hipStream_t stream) {
  (void)in_sizes; (void)n_in; (void)out_size; (void)ws_size;
  const float* x    = (const float*)d_in[0];  // [B,S,E]
  const float* Wqkv = (const float*)d_in[1];  // [H,3D,D]
  const float* Wout = (const float*)d_in[2];  // [E,E]
  float* out = (float*)d_out;                 // [B,S,E]

  _Float16* ws = (_Float16*)d_ws;
  const size_t BHSD = (size_t)B_ * H_ * S_ * D_;
  _Float16* hQ  = ws;                 // [BH,S,D]
  _Float16* hK  = hQ + BHSD;          // [BH,S,D]
  _Float16* hVt = hK + BHSD;          // [BH,D,S]
  _Float16* hO  = hVt + BHSD;         // [BH,S,D]
  _Float16* hWo = hO + BHSD;          // [E,E]
  _Float16* hWq = hWo + (size_t)E_ * E_;  // [H,3D,D]

  cvt_f32_to_f16<<<(E_ * E_ + 255) / 256, 256, 0, stream>>>(Wout, hWo, E_ * E_);
  cvt_f32_to_f16<<<(H_ * TD_ * D_ + 255) / 256, 256, 0, stream>>>(
      Wqkv, hWq, H_ * TD_ * D_);
  qkv_kernel<<<dim3(B_ * H_, S_ / 16), 32, 0, stream>>>(x, hWq, hQ, hK, hVt);
  attn_kernel<<<dim3(B_ * H_, S_ / 64), 128, 0, stream>>>(hQ, hK, hVt, hO);
  proj_kernel<<<dim3(B_ * S_ / 16, E_ / 64), 32, 0, stream>>>(hO, hWo, out);
}
